// AdditiveAttention_81724637708865
// MI455X (gfx1250) — matrix operand order repaired and checked by reading, not compiled, not run on hardware
//
#include <hip/hip_runtime.h>

typedef __attribute__((ext_vector_type(16))) _Float16 v16h;
typedef __attribute__((ext_vector_type(8)))  _Float16 v8h;
typedef __attribute__((ext_vector_type(4)))  _Float16 v4h;
typedef __attribute__((ext_vector_type(8)))  float    v8f;

#define GB 4
#define GLQ 256
#define GLK 256
#define GD 768

// ---------------- WMMA GEMM: C[M,N] = A[M,K] @ B[K,N] (+bias), fp32 IO, f16 WMMA ----------------
// 256 threads = 8 waves (wave32), block tile 64x128, BK=64, wave grid 2x4 (32x32/wave),
// double-buffered LDS software pipeline: fetch(i+1) || wmma(i) ; 1 barrier per K-step.
#define BM 64
#define BN 128
#define BKK 64
#define LP 72   // f16 LDS row pitch: 144B (16B-aligned v8h reads, bank-spread)
#define NA 4    // A float4 quads per thread per tile
#define NB 8    // B quads per thread per tile

__global__ __launch_bounds__(256) void gemm_wmma_f16(
    const float* __restrict__ A, long sA,
    const float* __restrict__ B, long sB,
    const float* __restrict__ bias,
    float* __restrict__ C, long sC,
    int M, int N, int K)
{
    __shared__ _Float16 lA[2][BM][LP];   // lA[buf][m][k]
    __shared__ _Float16 lB[2][BN][LP];   // lB[buf][n][k]  (B transposed in LDS)

    const int tid  = threadIdx.x;
    const int lane = tid & 31;
    const int wave = tid >> 5;
    const int wm   = wave & 1;   // 0..1
    const int wn   = wave >> 1;  // 0..3

    A += (long)blockIdx.z * sA;
    B += (long)blockIdx.z * sB;
    C += (long)blockIdx.z * sC;

    const int m_base = blockIdx.x * BM;
    const int n_base = blockIdx.y * BN;

    v8f acc[2][2];
#pragma unroll
    for (int tm = 0; tm < 2; ++tm)
#pragma unroll
        for (int tn = 0; tn < 2; ++tn)
#pragma unroll
            for (int e = 0; e < 8; ++e) acc[tm][tn][e] = 0.0f;

    const int mrow  = lane & 15;
    const int khalf = lane >> 4;   // 0 or 1

    float4 ra[NA];        // in-flight A tile
    float  rb[NB * 4];    // in-flight B tile

    // --- helpers ---
    auto fetch_tile = [&](int k0) {
#pragma unroll
        for (int i = 0; i < NA; ++i) {
            int e = i * 256 + tid;
            int r = e >> 4, qc = (e & 15) << 2;
            ra[i] = *(const float4*)(A + (long)(m_base + r) * K + (k0 + qc));
        }
#pragma unroll
        for (int i = 0; i < NB; ++i) {
            int e = i * 256 + tid;
            int n = e & 127, kq = (e >> 7) << 2;
            const float* pb = B + (long)(k0 + kq) * N + (n_base + n);
            rb[i * 4 + 0] = pb[0];
            rb[i * 4 + 1] = pb[(long)N];
            rb[i * 4 + 2] = pb[2L * N];
            rb[i * 4 + 3] = pb[3L * N];
        }
    };
    auto stage_tile = [&](int buf) {
#pragma unroll
        for (int i = 0; i < NA; ++i) {
            int e = i * 256 + tid;
            int r = e >> 4, qc = (e & 15) << 2;
            v4h h;
            h[0] = (_Float16)ra[i].x; h[1] = (_Float16)ra[i].y;
            h[2] = (_Float16)ra[i].z; h[3] = (_Float16)ra[i].w;
            *(v4h*)&lA[buf][r][qc] = h;
        }
#pragma unroll
        for (int i = 0; i < NB; ++i) {
            int e = i * 256 + tid;
            int n = e & 127, kq = (e >> 7) << 2;
            v4h h;
            h[0] = (_Float16)rb[i * 4 + 0]; h[1] = (_Float16)rb[i * 4 + 1];
            h[2] = (_Float16)rb[i * 4 + 2]; h[3] = (_Float16)rb[i * 4 + 3];
            *(v4h*)&lB[buf][n][kq] = h;
        }
    };

    // --- prologue: tile 0 ---
    fetch_tile(0);
    stage_tile(0);
    __syncthreads();

    int buf = 0;
    for (int k0 = 0; k0 < K; k0 += BKK) {
        const bool has_next = (k0 + BKK < K);
        if (has_next) fetch_tile(k0 + BKK);          // issue loads; wait deferred past wmma
        if (k0 + 2 * BKK < K) {                      // prefetch tile after next (128B lines)
            int r = tid >> 2, ch = (tid & 3) << 5;
            __builtin_prefetch(B + (long)(k0 + 2 * BKK + r) * N + (n_base + ch), 0, 3);
            if (tid < 128) {
                int r2 = tid >> 1, ch2 = (tid & 1) << 5;
                __builtin_prefetch(A + (long)(m_base + r2) * K + (k0 + 2 * BKK + ch2), 0, 3);
            }
        }

        // compute on current buffer: 2 k-steps x 4 wmma
#pragma unroll
        for (int ks = 0; ks < BKK; ks += 32) {
            v16h af[2], bf[2];
#pragma unroll
            for (int tm = 0; tm < 2; ++tm) {
                const _Float16* pa = &lA[buf][wm * 32 + tm * 16 + mrow][ks];
                v8h lo = *(const v8h*)(pa + khalf * 8);
                v8h hi = *(const v8h*)(pa + 16 + khalf * 8);
                af[tm] = __builtin_shufflevector(lo, hi, 0,1,2,3,4,5,6,7,8,9,10,11,12,13,14,15);
            }
#pragma unroll
            for (int tn = 0; tn < 2; ++tn) {
                const _Float16* pb = &lB[buf][wn * 32 + tn * 16 + mrow][ks];
                v8h lo = *(const v8h*)(pb + khalf * 8);
                v8h hi = *(const v8h*)(pb + 16 + khalf * 8);
                bf[tn] = __builtin_shufflevector(lo, hi, 0,1,2,3,4,5,6,7,8,9,10,11,12,13,14,15);
            }
#pragma unroll
            for (int tm = 0; tm < 2; ++tm)
#pragma unroll
                for (int tn = 0; tn < 2; ++tn)
                    acc[tm][tn] = __builtin_amdgcn_wmma_f32_16x16x32_f16(
                        false, af[tm], false, bf[tn], (short)0, acc[tm][tn], false, false);
        }

        if (has_next) stage_tile(buf ^ 1);           // cvt+ds_store next tile
        __syncthreads();
        buf ^= 1;
    }

    // Epilogue: bias hoisted, row-stride stores.
    const int n0 = n_base + wn * 32 + mrow;
    float badd[2] = {0.0f, 0.0f};
    if (bias) { badd[0] = bias[n0]; badd[1] = bias[n0 + 16]; }

#pragma unroll
    for (int tm = 0; tm < 2; ++tm) {
        const int m0 = m_base + wm * 32 + tm * 16 + (khalf << 3);
#pragma unroll
        for (int tn = 0; tn < 2; ++tn) {
            float* p = C + (long)m0 * N + (n0 + tn * 16);
#pragma unroll
            for (int r = 0; r < 8; ++r)
                p[(long)r * N] = acc[tm][tn][r] + badd[tn];
        }
    }
}

// ---------------- Scores + softmax ----------------
// One block per (b, 8-row q tile). 8 waves; each wave owns one key j per step; lanes stride D
// (coalesced, L2-resident k stream), 8 tanh+fma per loaded k element.
// softmax is shift-invariant => bs omitted (cancels exactly).
#define TQ 8
__global__ __launch_bounds__(256) void scores_softmax(
    const float* __restrict__ qp, const float* __restrict__ kp,
    const float* __restrict__ Ws, float* __restrict__ wout)
{
    __shared__ float qs[TQ][GD];     // 24 KB
    __shared__ float wsl[GD];        //  3 KB
    __shared__ float sc[TQ][GLK];    //  8 KB

    const int tid  = threadIdx.x;
    const int lane = tid & 31;
    const int wave = tid >> 5;
    const int b  = blockIdx.x >> 5;          // 32 q-tiles per batch
    const int q0 = (blockIdx.x & 31) * TQ;

#pragma unroll
    for (int qi = 0; qi < TQ; ++qi)
        for (int d = tid; d < GD; d += 256)
            qs[qi][d] = qp[(long)((b * GLQ) + q0 + qi) * GD + d];
    for (int d = tid; d < GD; d += 256) wsl[d] = Ws[d];
    __syncthreads();

    for (int j0 = 0; j0 < GLK; j0 += 8) {
        const int j = j0 + wave;
        const float* krow = kp + (long)(b * GLK + j) * GD;
        // Prefetch this wave's next key row (768 floats = 24 x 128B lines)
        if (j0 + 8 < GLK && lane < 24)
            __builtin_prefetch(krow + 8L * GD + (lane << 5), 0, 3);

        float acc[TQ];
#pragma unroll
        for (int qi = 0; qi < TQ; ++qi) acc[qi] = 0.0f;

        for (int d = lane; d < GD; d += 32) {
            float kd = krow[d];
            float wd = wsl[d];
#pragma unroll
            for (int qi = 0; qi < TQ; ++qi)
                acc[qi] += wd * tanhf(qs[qi][d] + kd);
        }
#pragma unroll
        for (int qi = 0; qi < TQ; ++qi) {
            float v = acc[qi];
            for (int off = 16; off; off >>= 1) v += __shfl_xor(v, off, 32);
            if (lane == qi) sc[qi][j] = v;
        }
    }
    __syncthreads();

    // softmax: wave w handles row w, 8 cols per lane
    {
        const int row = wave;
        float v[GLK / 32];
        float mx = -3.0e38f;
#pragma unroll
        for (int i = 0; i < GLK / 32; ++i) {
            v[i] = sc[row][lane + 32 * i];
            mx = fmaxf(mx, v[i]);
        }
        for (int off = 16; off; off >>= 1) mx = fmaxf(mx, __shfl_xor(mx, off, 32));
        float s = 0.0f;
#pragma unroll
        for (int i = 0; i < GLK / 32; ++i) {
            v[i] = __expf(v[i] - mx);
            s += v[i];
        }
        for (int off = 16; off; off >>= 1) s += __shfl_xor(s, off, 32);
        float inv = 1.0f / s;
        float* out = wout + (long)(b * GLQ + q0 + row) * GLK;
#pragma unroll
        for (int i = 0; i < GLK / 32; ++i) out[lane + 32 * i] = v[i] * inv;
    }
}

extern "C" void kernel_launch(void* const* d_in, const int* in_sizes, int n_in,
                              void* d_out, int out_size, void* d_ws, size_t ws_size,
                              hipStream_t stream) {
    const float* query = (const float*)d_in[0];
    const float* key   = (const float*)d_in[1];
    const float* value = (const float*)d_in[2];
    const float* Wq    = (const float*)d_in[3];
    const float* bq    = (const float*)d_in[4];
    const float* Wk    = (const float*)d_in[5];
    const float* bk    = (const float*)d_in[6];
    const float* Wv    = (const float*)d_in[7];
    const float* bv    = (const float*)d_in[8];
    const float* Ws    = (const float*)d_in[9];
    // d_in[10] = bs : softmax is shift-invariant, bs cancels exactly.

    const int Mproj = GB * GLQ;                 // 1024
    float* qp = (float*)d_ws;                   // [1024,768]
    float* kp = qp + (long)Mproj * GD;
    float* vp = kp + (long)Mproj * GD;

    float* attended = (float*)d_out;                         // [4,256,768]
    float* weights  = attended + (long)GB * GLQ * GD;        // [4,256,256]

    // 1) Projections: X @ W + b  (M=1024, N=K=768)
    dim3 gP(Mproj / BM, GD / BN, 1);
    gemm_wmma_f16<<<gP, 256, 0, stream>>>(query, 0, Wq, 0, bq, qp, 0, Mproj, GD, GD);
    gemm_wmma_f16<<<gP, 256, 0, stream>>>(key,   0, Wk, 0, bk, kp, 0, Mproj, GD, GD);
    gemm_wmma_f16<<<gP, 256, 0, stream>>>(value, 0, Wv, 0, bv, vp, 0, Mproj, GD, GD);

    // 2) Additive scores + softmax -> attention weights (second output)
    scores_softmax<<<GB * (GLQ / TQ), 256, 0, stream>>>(qp, kp, Ws, weights);

    // 3) attended = weights @ v   (batched: M=256, K=256, N=768)
    dim3 gAV(GLQ / BM, GD / BN, GB);
    gemm_wmma_f16<<<gAV, 256, 0, stream>>>(weights, (long)GLQ * GLK,
                                           vp,      (long)GLK * GD,
                                           nullptr,
                                           attended,(long)GLQ * GD,
                                           GLQ, GD, GLK);
}
